// LatticeMamba_27874337751279
// MI455X (gfx1250) — matrix operand order repaired and checked by reading, not compiled, not run on hardware
//
#include <hip/hip_runtime.h>
#include <hip/hip_fp16.h>
#include <math.h>

#define B_ 8
#define L_ 1024
#define DM_ 256
#define NL_ 4
#define ED_ 512
#define NS_ 16
#define DC_ 4
#define RK_ 16
#define CD_ 64
#define Q_ 2
#define ROWS_ (B_ * L_)      /* 8192 */
#define XPN_ (RK_ + 2 * NS_) /* 48 */

typedef __attribute__((ext_vector_type(16))) _Float16 v16h;
typedef __attribute__((ext_vector_type(8)))  float    v8f;

union HVec { uint4 q[2]; v16h v; };

// ---------------------------------------------------------------------------
// WMMA GEMM: C[M,N] (f32) = A[M,K] (f16 row-major) * Bt[N,K]^T (f16, weights
// pre-transposed so each lane's K-fragment is two contiguous 16B loads).
// Each wave computes a 16 x (16*NT) output strip.  All NT B fragments are
// loaded BEFORE the WMMA sequence so the compiler can emit one long load
// clause and stagger s_wait_loadcnt across the NT v_wmma ops (latency
// overlap instead of full-drain waits).
// Fragment layouts per CDNA5 ISA tables:
//   A (16x32): g=lane>>4, m=lane&15; j<8 -> k=k0+g*8+j, j>=8 -> k=k0+16+g*8+j-8
//   B (32x16): n=lane&15, same k mapping as A (two 16B loads per lane)
//   D: vgpr r -> row tm+r+g*8, col tn+(lane&15)
// ---------------------------------------------------------------------------
template <int ACCUM, int NT>
__global__ __launch_bounds__(256) void wmma_gemm_kernel(
    const _Float16* __restrict__ A, const _Float16* __restrict__ Bt,
    float* __restrict__ C, int M, int N, int K) {
  const int lane = threadIdx.x & 31;
  const int wave = threadIdx.x >> 5;
  const int g = lane >> 4;
  const int mr = lane & 15;
  const int tilesN = N / (16 * NT);
  const int totalTiles = (M >> 4) * tilesN;
  const int t = blockIdx.x * 8 + wave;
  if (t >= totalTiles) return;  // wave-uniform: EXEC stays all-ones for WMMA
  const int tm = (t / tilesN) << 4;
  const int tn = (t % tilesN) * (16 * NT);

  v8f acc[NT];
#pragma unroll
  for (int j = 0; j < NT; ++j) acc[j] = (v8f){};

  const _Float16* arow = A + (size_t)(tm + mr) * K;
  const _Float16* brow[NT];
#pragma unroll
  for (int j = 0; j < NT; ++j)
    brow[j] = Bt + (size_t)(tn + j * 16 + mr) * K;

  for (int k0 = 0; k0 < K; k0 += 32) {
    HVec a, b[NT];
    // Issue ALL loads for this K-step first (one long clause).
    a.q[0] = *(const uint4*)(arow + k0 + g * 8);
    a.q[1] = *(const uint4*)(arow + k0 + 16 + g * 8);
#pragma unroll
    for (int j = 0; j < NT; ++j) {
      b[j].q[0] = *(const uint4*)(brow[j] + k0 + g * 8);
      b[j].q[1] = *(const uint4*)(brow[j] + k0 + 16 + g * 8);
    }
    // Then the NT matrix ops: waits can retire incrementally.
#pragma unroll
    for (int j = 0; j < NT; ++j)
      acc[j] = __builtin_amdgcn_wmma_f32_16x16x32_f16(
          false, a.v, false, b[j].v, (short)0, acc[j], false, false);
  }
#pragma unroll
  for (int j = 0; j < NT; ++j) {
#pragma unroll
    for (int r = 0; r < 8; ++r) {
      size_t idx = (size_t)(tm + r + g * 8) * N + tn + j * 16 + mr;
      if (ACCUM) C[idx] += acc[j][r];
      else       C[idx] = acc[j][r];
    }
  }
}

// Convert + transpose weights to f16: W[R,C] (f32) -> Wt[C,R] (f16)
__global__ void convt_kernel(const float* __restrict__ W,
                             _Float16* __restrict__ Wt, int R, int C) {
  int i = blockIdx.x * 256 + threadIdx.x;
  if (i >= R * C) return;
  int r = i / C, c = i % C;
  Wt[(size_t)c * R + r] = (_Float16)W[i];
}

// x[b,l,:] = (l>0 ? tok_embed[tokens[b,l-1]] : 0) + pos_embed[l]
__global__ void embed_kernel(const int* __restrict__ tokens,
                             const float* __restrict__ tok_embed,
                             const float* __restrict__ pos_embed,
                             float* __restrict__ x) {
  int i = blockIdx.x * 256 + threadIdx.x;
  if (i >= ROWS_ * DM_) return;
  int row = i / DM_, d = i % DM_;
  int b = row / L_, l = row % L_;
  float v = pos_embed[l * DM_ + d];
  if (l > 0) v += tok_embed[tokens[b * L_ + l - 1] * DM_ + d];
  x[i] = v;
}

// cond = gelu(T@tw1+tb1)@tw2+tb2 ; ss[i,b,:] = cond[b]@apw[i]+apb[i]
__global__ __launch_bounds__(256) void cond_kernel(
    const float* __restrict__ T, const float* __restrict__ tw1,
    const float* __restrict__ tb1, const float* __restrict__ tw2,
    const float* __restrict__ tb2, const float* __restrict__ apw,
    const float* __restrict__ apb, float* __restrict__ ss) {
  __shared__ float sh_h[B_][CD_];
  __shared__ float sh_c[B_][CD_];
  int tid = threadIdx.x;
  for (int i = tid; i < B_ * CD_; i += 256) {
    int b = i / CD_, c = i % CD_;
    float xv = T[b] * tw1[c] + tb1[c];
    sh_h[b][c] = 0.5f * xv * (1.0f + erff(xv * 0.70710678118654752f));
  }
  __syncthreads();
  for (int i = tid; i < B_ * CD_; i += 256) {
    int b = i / CD_, d = i % CD_;
    float s = tb2[d];
    for (int c = 0; c < CD_; ++c) s += sh_h[b][c] * tw2[c * CD_ + d];
    sh_c[b][d] = s;
  }
  __syncthreads();
  for (int i = tid; i < NL_ * B_ * 2 * DM_; i += 256) {
    int il = i / (B_ * 2 * DM_);
    int rem = i % (B_ * 2 * DM_);
    int b = rem / (2 * DM_), j = rem % (2 * DM_);
    const float* w = apw + (size_t)il * CD_ * 2 * DM_;
    float s = apb[il * 2 * DM_ + j];
    for (int c = 0; c < CD_; ++c) s += sh_c[b][c] * w[c * 2 * DM_ + j];
    ss[i] = s;
  }
}

// Adaptive LN: xn16[row,:] = (1+sc)*LN(x)*g+b + sh  (one block per row)
__global__ __launch_bounds__(256) void adaln_kernel(
    const float* __restrict__ x, const float* __restrict__ ss_l,
    const float* __restrict__ g, const float* __restrict__ bb,
    _Float16* __restrict__ xn16) {
  __shared__ float red[256];
  int row = blockIdx.x;
  int tid = threadIdx.x;
  int b = row / L_;
  float v = x[(size_t)row * DM_ + tid];
  red[tid] = v;
  __syncthreads();
  for (int s = 128; s > 0; s >>= 1) {
    if (tid < s) red[tid] += red[tid + s];
    __syncthreads();
  }
  float mean = red[0] * (1.0f / DM_);
  __syncthreads();
  float d = v - mean;
  red[tid] = d * d;
  __syncthreads();
  for (int s = 128; s > 0; s >>= 1) {
    if (tid < s) red[tid] += red[tid + s];
    __syncthreads();
  }
  float var = red[0] * (1.0f / DM_);
  float nrm = d * rsqrtf(var + 1e-5f) * g[tid] + bb[tid];
  float sc = ss_l[b * 2 * DM_ + tid];
  float shv = ss_l[b * 2 * DM_ + DM_ + tid];
  xn16[(size_t)row * DM_ + tid] = (_Float16)((1.0f + sc) * nrm + shv);
}

// Depthwise causal conv (DC=4) + bias + silu. xs part of xz is cols [0,ED).
__global__ void conv_kernel(const float* __restrict__ xz,
                            const float* __restrict__ cw,
                            const float* __restrict__ cb,
                            float* __restrict__ xs32,
                            _Float16* __restrict__ xs16) {
  int i = blockIdx.x * 256 + threadIdx.x;
  if (i >= ROWS_ * ED_) return;
  int row = i / ED_, e = i % ED_;
  int l = row % L_;
  float s = cb[e];
#pragma unroll
  for (int d = 0; d < DC_; ++d) {
    int ls = l - (DC_ - 1) + d;
    if (ls >= 0)
      s += xz[(size_t)(row - (DC_ - 1) + d) * (2 * ED_) + e] * cw[e * DC_ + d];
  }
  float sl = s / (1.0f + __expf(-s));
  xs32[i] = sl;
  xs16[i] = (_Float16)sl;
}

// dlt = softplus(xd[:, :RK] @ dt_w + dt_b)   grid (ROWS, ED/256)
__global__ __launch_bounds__(256) void dlt_kernel(
    const float* __restrict__ xd, const float* __restrict__ dtw,
    const float* __restrict__ dtb, float* __restrict__ dlt) {
  __shared__ float sx[RK_];
  int row = blockIdx.x;
  int tid = threadIdx.x;
  if (tid < RK_) sx[tid] = xd[(size_t)row * XPN_ + tid];
  __syncthreads();
  int e = blockIdx.y * 256 + tid;
  float s = dtb[e];
#pragma unroll
  for (int r = 0; r < RK_; ++r) s += sx[r] * dtw[r * ED_ + e];
  float sp = (s > 0.0f) ? (s + log1pf(__expf(-s))) : log1pf(__expf(s));
  dlt[(size_t)row * ED_ + e] = sp;
}

// Sequential selective scan: h[n] = exp(dlt*A[e,n])*h[n] + dlt*xs*Bm[n];
// y = sum_n h[n]*Cm[n] + Dp[e]*xs.  One thread per (b,e); Bm/Cm staged in LDS.
__global__ __launch_bounds__(256) void scan_kernel(
    const float* __restrict__ dlt, const float* __restrict__ xs32,
    const float* __restrict__ xd, const float* __restrict__ Alog,
    const float* __restrict__ Dp, float* __restrict__ y) {
  __shared__ float sB[NS_], sC[NS_];
  int b = blockIdx.x;
  int tid = threadIdx.x;
  int e = blockIdx.y * 256 + tid;
  float a_c[NS_], h[NS_];
#pragma unroll
  for (int n = 0; n < NS_; ++n) {
    a_c[n] = -__expf(Alog[e * NS_ + n]);
    h[n] = 0.0f;
  }
  float dpe = Dp[e];
  for (int l = 0; l < L_; ++l) {
    int row = b * L_ + l;
    if (tid < NS_) sB[tid] = xd[(size_t)row * XPN_ + RK_ + tid];
    else if (tid < 2 * NS_) sC[tid - NS_] = xd[(size_t)row * XPN_ + RK_ + tid];
    __syncthreads();
    float dv = dlt[(size_t)row * ED_ + e];
    float xv = xs32[(size_t)row * ED_ + e];
    float dx = dv * xv;
    float acc = 0.0f;
#pragma unroll
    for (int n = 0; n < NS_; ++n) {
      float da = __expf(dv * a_c[n]);
      h[n] = da * h[n] + dx * sB[n];
      acc += h[n] * sC[n];
    }
    y[(size_t)row * ED_ + e] = acc + dpe * xv;
    __syncthreads();
  }
}

// yg16 = (f16)(y * silu(z)); z part of xz is cols [ED, 2*ED)
__global__ void gate_kernel(const float* __restrict__ y,
                            const float* __restrict__ xz,
                            _Float16* __restrict__ yg16) {
  int i = blockIdx.x * 256 + threadIdx.x;
  if (i >= ROWS_ * ED_) return;
  int row = i / ED_, e = i % ED_;
  float z = xz[(size_t)row * (2 * ED_) + ED_ + e];
  float sz = z / (1.0f + __expf(-z));
  yg16[i] = (_Float16)(y[i] * sz);
}

// Final LN + head (N=2).  One block per row.
__global__ __launch_bounds__(256) void head_kernel(
    const float* __restrict__ x, const float* __restrict__ fg,
    const float* __restrict__ fb, const float* __restrict__ hw,
    const float* __restrict__ hb, float* __restrict__ out) {
  __shared__ float red[256];
  __shared__ float red2[256];
  int row = blockIdx.x, tid = threadIdx.x;
  float v = x[(size_t)row * DM_ + tid];
  red[tid] = v;
  __syncthreads();
  for (int s = 128; s > 0; s >>= 1) {
    if (tid < s) red[tid] += red[tid + s];
    __syncthreads();
  }
  float mean = red[0] * (1.0f / DM_);
  __syncthreads();
  float d = v - mean;
  red[tid] = d * d;
  __syncthreads();
  for (int s = 128; s > 0; s >>= 1) {
    if (tid < s) red[tid] += red[tid + s];
    __syncthreads();
  }
  float var = red[0] * (1.0f / DM_);
  float nrm = d * rsqrtf(var + 1e-5f) * fg[tid] + fb[tid];
  __syncthreads();
  red[tid] = nrm * hw[tid * Q_ + 0];
  red2[tid] = nrm * hw[tid * Q_ + 1];
  __syncthreads();
  for (int s = 128; s > 0; s >>= 1) {
    if (tid < s) {
      red[tid] += red[tid + s];
      red2[tid] += red2[tid + s];
    }
    __syncthreads();
  }
  if (tid == 0) {
    out[row * Q_ + 0] = red[0] + hb[0];
    out[row * Q_ + 1] = red2[0] + hb[1];
  }
}

extern "C" void kernel_launch(void* const* d_in, const int* in_sizes, int n_in,
                              void* d_out, int out_size, void* d_ws,
                              size_t ws_size, hipStream_t stream) {
  const int*   tokens    = (const int*)d_in[0];
  const float* T         = (const float*)d_in[1];
  const float* tok_embed = (const float*)d_in[2];
  const float* pos_embed = (const float*)d_in[3];
  const float* tw1       = (const float*)d_in[4];
  const float* tb1       = (const float*)d_in[5];
  const float* tw2       = (const float*)d_in[6];
  const float* tb2       = (const float*)d_in[7];
  const float* ag        = (const float*)d_in[8];
  const float* ab        = (const float*)d_in[9];
  const float* apw       = (const float*)d_in[10];
  const float* apb       = (const float*)d_in[11];
  const float* in_w      = (const float*)d_in[12];
  const float* conv_w    = (const float*)d_in[13];
  const float* conv_b    = (const float*)d_in[14];
  const float* xp_w      = (const float*)d_in[15];
  const float* dt_w      = (const float*)d_in[16];
  const float* dt_b      = (const float*)d_in[17];
  const float* A_log     = (const float*)d_in[18];
  const float* Dp        = (const float*)d_in[19];
  const float* out_w     = (const float*)d_in[20];
  const float* fg        = (const float*)d_in[21];
  const float* fb        = (const float*)d_in[22];
  const float* hw        = (const float*)d_in[23];
  const float* hb        = (const float*)d_in[24];
  float* outp = (float*)d_out;

  char* ws = (char*)d_ws;
  size_t off = 0;
  auto alloc = [&](size_t bytes) -> void* {
    off = (off + 255) & ~(size_t)255;
    void* p = ws + off;
    off += bytes;
    return p;
  };
  float*     x    = (float*)alloc((size_t)ROWS_ * DM_ * 4);
  _Float16*  xn16 = (_Float16*)alloc((size_t)ROWS_ * DM_ * 2);
  float*     xz   = (float*)alloc((size_t)ROWS_ * 2 * ED_ * 4);
  float*     xs32 = (float*)alloc((size_t)ROWS_ * ED_ * 4);
  _Float16*  xs16 = (_Float16*)alloc((size_t)ROWS_ * ED_ * 2);
  float*     xd   = (float*)alloc((size_t)ROWS_ * XPN_ * 4);
  float*     dlt  = (float*)alloc((size_t)ROWS_ * ED_ * 4);
  float*     ybuf = (float*)alloc((size_t)ROWS_ * ED_ * 4);
  _Float16*  yg16 = (_Float16*)alloc((size_t)ROWS_ * ED_ * 2);
  float*     ss   = (float*)alloc((size_t)NL_ * B_ * 2 * DM_ * 4);
  _Float16*  wt1  = (_Float16*)alloc((size_t)NL_ * 2 * ED_ * DM_ * 2);
  _Float16*  wtx  = (_Float16*)alloc((size_t)NL_ * XPN_ * ED_ * 2);
  _Float16*  wto  = (_Float16*)alloc((size_t)NL_ * DM_ * ED_ * 2);

  // Embedding + conditioning
  embed_kernel<<<(ROWS_ * DM_ + 255) / 256, 256, 0, stream>>>(tokens, tok_embed,
                                                              pos_embed, x);
  cond_kernel<<<1, 256, 0, stream>>>(T, tw1, tb1, tw2, tb2, apw, apb, ss);

  // Weight conversion (f32 -> transposed f16), all layers
  for (int i = 0; i < NL_; ++i) {
    convt_kernel<<<(DM_ * 2 * ED_ + 255) / 256, 256, 0, stream>>>(
        in_w + (size_t)i * DM_ * 2 * ED_, wt1 + (size_t)i * 2 * ED_ * DM_, DM_,
        2 * ED_);
    convt_kernel<<<(ED_ * XPN_ + 255) / 256, 256, 0, stream>>>(
        xp_w + (size_t)i * ED_ * XPN_, wtx + (size_t)i * XPN_ * ED_, ED_, XPN_);
    convt_kernel<<<(ED_ * DM_ + 255) / 256, 256, 0, stream>>>(
        out_w + (size_t)i * ED_ * DM_, wto + (size_t)i * DM_ * ED_, ED_, DM_);
  }

  for (int i = 0; i < NL_; ++i) {
    adaln_kernel<<<ROWS_, 256, 0, stream>>>(
        x, ss + (size_t)i * B_ * 2 * DM_, ag + (size_t)i * DM_,
        ab + (size_t)i * DM_, xn16);
    // xz = xn @ in_w   (M=8192, N=1024, K=256): 16x64 strip per wave
    wmma_gemm_kernel<0, 4>
        <<<(ROWS_ / 16) * (2 * ED_ / 64) / 8, 256, 0, stream>>>(
            xn16, wt1 + (size_t)i * 2 * ED_ * DM_, xz, ROWS_, 2 * ED_, DM_);
    conv_kernel<<<(ROWS_ * ED_ + 255) / 256, 256, 0, stream>>>(
        xz, conv_w + (size_t)i * ED_ * DC_, conv_b + (size_t)i * ED_, xs32,
        xs16);
    // xd = xs @ xp_w   (M=8192, N=48, K=512): single tile per wave
    wmma_gemm_kernel<0, 1>
        <<<(ROWS_ / 16) * (XPN_ / 16) / 8, 256, 0, stream>>>(
            xs16, wtx + (size_t)i * XPN_ * ED_, xd, ROWS_, XPN_, ED_);
    dlt_kernel<<<dim3(ROWS_, ED_ / 256), 256, 0, stream>>>(
        xd, dt_w + (size_t)i * RK_ * ED_, dt_b + (size_t)i * ED_, dlt);
    scan_kernel<<<dim3(B_, ED_ / 256), 256, 0, stream>>>(
        dlt, xs32, xd, A_log + (size_t)i * ED_ * NS_, Dp + (size_t)i * ED_,
        ybuf);
    gate_kernel<<<(ROWS_ * ED_ + 255) / 256, 256, 0, stream>>>(ybuf, xz, yg16);
    // x += yg @ out_w  (M=8192, N=256, K=512), 16x64 strip, accumulating
    wmma_gemm_kernel<1, 4>
        <<<(ROWS_ / 16) * (DM_ / 64) / 8, 256, 0, stream>>>(
            yg16, wto + (size_t)i * DM_ * ED_, x, ROWS_, DM_, ED_);
  }

  head_kernel<<<ROWS_, 256, 0, stream>>>(x, fg, fb, hw, hb, outp);
}
